// PKEModel_14328010900134
// MI455X (gfx1250) — compile-verified, matrix operand order checked by reading, not run on hardware
//
#include <hip/hip_runtime.h>
#include <hip/hip_bf16.h>

typedef __attribute__((ext_vector_type(16))) _Float16 v16h;
typedef __attribute__((ext_vector_type(8)))  float    v8f;

#define HID 64

// ---------------------------------------------------------------------------
// WMMA helper: D = A(16x32 f16) * B(32x16 f16) + C(16x16 f32)
// ---------------------------------------------------------------------------
__device__ inline v8f wmma_f16(v16h a, v16h b, v8f c) {
  return __builtin_amdgcn_wmma_f32_16x16x32_f16(false, a, false, b, (short)0, c,
                                                false, false);
}

// A fragment (16x32) from one f32 row (this lane's row), K window [k0,k0+32).
// ISA layout: lanes 0-15 halves 0..7 -> K=k0+0..7, halves 8..15 -> K=k0+16..23;
// lanes 16-31 offset +8.  No bounds checks: caller guarantees k0+24+7 in range.
__device__ inline v16h load_a_row(const float* __restrict__ row, int k0) {
  const int lane = threadIdx.x & 31;
  const int kb = k0 + ((lane & 16) ? 8 : 0);
  v16h a;
#pragma unroll
  for (int i = 0; i < 8; ++i) {
    a[i]     = (_Float16)row[kb + i];
    a[i + 8] = (_Float16)row[kb + 16 + i];
  }
  return a;
}

// K=16 variant: upper half of the 32-wide K window is statically zero.
__device__ inline v16h load_a_row_k16(const float* __restrict__ row) {
  const int lane = threadIdx.x & 31;
  const int kb = (lane & 16) ? 8 : 0;
  v16h a;
#pragma unroll
  for (int i = 0; i < 8; ++i) {
    a[i]     = (_Float16)row[kb + i];
    a[i + 8] = (_Float16)0.0f;
  }
  return a;
}

// A fragment from an LDS-resident f16 tile [16][64].
__device__ inline v16h load_a_lds(const _Float16* __restrict__ t, int k0) {
  const int lane = threadIdx.x & 31;
  const int m = lane & 15;
  const int kb = k0 + ((lane & 16) ? 8 : 0);
  v16h a;
#pragma unroll
  for (int i = 0; i < 8; ++i) {
    a[i]     = t[m * HID + kb + i];
    a[i + 8] = t[m * HID + kb + 16 + i];
  }
  return a;
}

// B fragment from pre-packed weights: one contiguous 32B load per lane.
// Packed layout: [frag = kt*4+nt][lane 0..31][16 halves]
__device__ inline v16h load_b_packed(const _Float16* __restrict__ Wp, int frag) {
  const int lane = threadIdx.x & 31;
  return *(const v16h*)(Wp + (((size_t)frag * 32 + lane) << 4));
}

// ---------------------------------------------------------------------------
// Utility kernels
// ---------------------------------------------------------------------------

// Pack f32 weight [K][64] into B-fragment-ordered f16 (zero-pad K to mult of 32).
__global__ void k_pack_b(const float* __restrict__ src, _Float16* __restrict__ dstP,
                         int K) {
  const int tid = blockIdx.x * blockDim.x + threadIdx.x;
  const int KT = (K + 31) >> 5;
  const int total = KT * 4 * 32 * 16;
  if (tid >= total) return;
  const int i = tid & 15;
  const int lane = (tid >> 4) & 31;
  const int f = tid >> 9;  // kt*4 + nt
  const int nt = f & 3;
  const int kt = f >> 2;
  const int kb = (lane & 16) ? 8 : 0;
  const int k = kt * 32 + kb + ((i < 8) ? i : (i - 8) + 16);
  const int n = nt * 16 + (lane & 15);
  dstP[tid] = (k < K) ? (_Float16)src[(size_t)k * HID + n] : (_Float16)0.0f;
}

__global__ void k_f32_to_f16(const float* __restrict__ s, _Float16* __restrict__ d,
                             int n) {
  int i = blockIdx.x * blockDim.x + threadIdx.x;
  if (i < n) d[i] = (_Float16)s[i];
}

__global__ void k_fill_f32(float* __restrict__ p, float v, long long n) {
  long long i = (long long)blockIdx.x * blockDim.x + threadIdx.x;
  long long stride = (long long)gridDim.x * blockDim.x;
  for (; i < n; i += stride) p[i] = v;
}

__device__ inline void atomicMaxF32(float* addr, float val) {
  unsigned int* a = (unsigned int*)addr;
  unsigned int old = *a;
  while (__uint_as_float(old) < val) {
    unsigned int assumed = old;
    old = atomicCAS(a, assumed, __float_as_uint(val));
    if (old == assumed) break;
  }
}

// ---------------------------------------------------------------------------
// Node GEMM: out[M x 64] = epilogue(A[M x K] @ W[K x 64] + bias)
// MODE 0: +bias   MODE 1: relu(+bias)   MODE 2: resid + relu(+bias)
// one wave = 16 rows x 64 cols (4 col tiles); K is compile-time (16 or 64).
// ---------------------------------------------------------------------------
template <int K, int MODE>
__global__ void k_gemm_node(const float* __restrict__ A,
                            const _Float16* __restrict__ Wp,
                            const float* __restrict__ bias,
                            const float* __restrict__ resid,
                            float* __restrict__ out, int M) {
  constexpr int NK = (K + 31) / 32;
  const int wave = threadIdx.x >> 5;
  const int lane = threadIdx.x & 31;
  const int rt = blockIdx.x * (blockDim.x >> 5) + wave;
  const int row0 = rt * 16;
  if (row0 >= M) return;

  const float* rowp = A + (size_t)(row0 + (lane & 15)) * K;
  v16h afr[NK];
  if constexpr (K == 16) {
    afr[0] = load_a_row_k16(rowp);
  } else {
#pragma unroll
    for (int kk = 0; kk < NK; ++kk) afr[kk] = load_a_row(rowp, kk * 32);
  }

  const int nlo = lane & 15;
  const int mhi = (lane & 16) ? 8 : 0;

#pragma unroll
  for (int nt = 0; nt < 4; ++nt) {
    v8f c = {0.f, 0.f, 0.f, 0.f, 0.f, 0.f, 0.f, 0.f};
#pragma unroll
    for (int kk = 0; kk < NK; ++kk)
      c = wmma_f16(afr[kk], load_b_packed(Wp, kk * 4 + nt), c);
    const int n = nt * 16 + nlo;
    const float bn = bias[n];
#pragma unroll
    for (int v = 0; v < 8; ++v) {
      const int m = row0 + mhi + v;
      float val = c[v] + bn;
      if constexpr (MODE >= 1) val = fmaxf(val, 0.f);
      if constexpr (MODE == 2) val += resid[(size_t)m * HID + n];
      out[(size_t)m * HID + n] = val;
    }
  }
}

// ---------------------------------------------------------------------------
// GATv2 edge passes (thread per edge*head; self-loops are edges E..E+N-1)
// ---------------------------------------------------------------------------
__global__ void k_gat_p1(const float* __restrict__ gl, const float* __restrict__ gr,
                         const int* __restrict__ src, const int* __restrict__ dst,
                         const float* __restrict__ att, float* __restrict__ ebuf,
                         float* __restrict__ mx, int E, int EP) {
  int t = blockIdx.x * blockDim.x + threadIdx.x;
  if (t >= EP * 4) return;
  int e = t >> 2, hh = t & 3;
  int s, d;
  if (e < E) { s = src[e]; d = dst[e]; } else { s = d = e - E; }
  const float* glp = gl + (size_t)s * HID + hh * 16;
  const float* grp = gr + (size_t)d * HID + hh * 16;
  const float* ap  = att + hh * 16;
  float acc = 0.f;
#pragma unroll
  for (int i = 0; i < 16; ++i) {
    float v = glp[i] + grp[i];
    v = (v > 0.f) ? v : 0.2f * v;  // leaky_relu(0.2)
    acc += ap[i] * v;
  }
  ebuf[t] = acc;
  atomicMaxF32(&mx[d * 4 + hh], acc);
}

__global__ void k_gat_p2(float* __restrict__ ebuf, const int* __restrict__ dst,
                         const float* __restrict__ mx, float* __restrict__ sm,
                         int E, int EP) {
  int t = blockIdx.x * blockDim.x + threadIdx.x;
  if (t >= EP * 4) return;
  int e = t >> 2, hh = t & 3;
  int d = (e < E) ? dst[e] : (e - E);
  float a = __expf(ebuf[t] - mx[d * 4 + hh]);
  ebuf[t] = a;
  atomicAdd(&sm[d * 4 + hh], a);
}

__global__ void k_gat_p3(const float* __restrict__ ebuf, const float* __restrict__ gl,
                         const int* __restrict__ src, const int* __restrict__ dst,
                         const float* __restrict__ sm, float* __restrict__ acc,
                         int E, int EP) {
  int t = blockIdx.x * blockDim.x + threadIdx.x;
  if (t >= EP * 4) return;
  int e = t >> 2, hh = t & 3;
  int s, d;
  if (e < E) { s = src[e]; d = dst[e]; } else { s = d = e - E; }
  float alpha = ebuf[t] / sm[d * 4 + hh];
  const float* glp = gl + (size_t)s * HID + hh * 16;
  float* ap = acc + (size_t)d * HID + hh * 16;
#pragma unroll
  for (int i = 0; i < 16; ++i) atomicAdd(&ap[i], alpha * glp[i]);
}

__global__ void k_gat_fin(const float* __restrict__ hIn, const float* __restrict__ acc,
                          const float* __restrict__ cb, float* __restrict__ hOut,
                          int n) {
  int i = blockIdx.x * blockDim.x + threadIdx.x;
  if (i < n) hOut[i] = hIn[i] + acc[i] + cb[i & (HID - 1)];
}

// ---------------------------------------------------------------------------
// Fused edge MLP + scatter:
//   ef1 = concat(h[src],h[dst]) @ We1 + be1          (no activation)
//   ef2 = relu(ef1 @ We2 + be2)
//   ev  = sigmoid(ef2 . We3 + be3)
//   out[src*N + dst] = ev
// One wave per 16-edge tile; intermediates staged through LDS (f16).
// ---------------------------------------------------------------------------
__global__ void k_edge_mlp(const float* __restrict__ h,
                           const int* __restrict__ src, const int* __restrict__ dst,
                           const _Float16* __restrict__ We1p,  // packed, KT=4
                           const float* __restrict__ be1,
                           const _Float16* __restrict__ We2p,  // packed, KT=2
                           const float* __restrict__ be2,
                           const _Float16* __restrict__ We3,   // flat [64]
                           const float* __restrict__ be3,
                           float* __restrict__ out, int N, int E) {
  __shared__ _Float16 tiles[8][16 * HID];
  const int wave = threadIdx.x >> 5;
  const int lane = threadIdx.x & 31;
  int e0 = (blockIdx.x * (blockDim.x >> 5) + wave) * 16;
  if (e0 > E - 16) e0 = E - 16;  // clamp: duplicate tiles rewrite identical values
  _Float16* tile = tiles[wave];

  const int erow = e0 + (lane & 15);
  const float* srow = h + (size_t)src[erow] * HID;
  const float* drow = h + (size_t)dst[erow] * HID;
  const int nlo = lane & 15;
  const int mhi = (lane & 16) ? 8 : 0;

  // ---- stage 1: [16x128] @ We1 + be1 -> LDS (f16) ----
  // K windows 0/32 read h[src], 64/96 read h[dst] (branch resolved per window).
  v16h a1[4];
  a1[0] = load_a_row(srow, 0);
  a1[1] = load_a_row(srow, 32);
  a1[2] = load_a_row(drow, 0);
  a1[3] = load_a_row(drow, 32);

#pragma unroll
  for (int nt = 0; nt < 4; ++nt) {
    v8f c = {0.f, 0.f, 0.f, 0.f, 0.f, 0.f, 0.f, 0.f};
#pragma unroll
    for (int kk = 0; kk < 4; ++kk)
      c = wmma_f16(a1[kk], load_b_packed(We1p, kk * 4 + nt), c);
    const int n = nt * 16 + nlo;
    const float bn = be1[n];
#pragma unroll
    for (int v = 0; v < 8; ++v)
      tile[(mhi + v) * HID + n] = (_Float16)(c[v] + bn);
  }
  __syncthreads();

  // ---- stage 2: relu([16x64] @ We2 + be2) -> LDS (f16, overwrite) ----
  v16h a2[2];
  a2[0] = load_a_lds(tile, 0);
  a2[1] = load_a_lds(tile, 32);
  __syncthreads();

#pragma unroll
  for (int nt = 0; nt < 4; ++nt) {
    v8f c = {0.f, 0.f, 0.f, 0.f, 0.f, 0.f, 0.f, 0.f};
    c = wmma_f16(a2[0], load_b_packed(We2p, 0 * 4 + nt), c);
    c = wmma_f16(a2[1], load_b_packed(We2p, 1 * 4 + nt), c);
    const int n = nt * 16 + nlo;
    const float bn = be2[n];
#pragma unroll
    for (int v = 0; v < 8; ++v)
      tile[(mhi + v) * HID + n] = (_Float16)fmaxf(c[v] + bn, 0.f);
  }
  __syncthreads();

  // ---- stage 3: per-edge dot with We3, sigmoid, scatter ----
  if (lane < 16) {
    const int e = e0 + lane;
    float d = 0.f;
#pragma unroll
    for (int k = 0; k < HID; ++k)
      d += (float)tile[lane * HID + k] * (float)We3[k];
    d += be3[0];
    float ev = 1.f / (1.f + __expf(-d));
    out[(size_t)src[e] * N + dst[e]] = ev;
  }
}

// ---------------------------------------------------------------------------
// Host orchestration
// ---------------------------------------------------------------------------
extern "C" void kernel_launch(void* const* d_in, const int* in_sizes, int n_in,
                              void* d_out, int out_size, void* d_ws, size_t ws_size,
                              hipStream_t stream) {
  const int N = in_sizes[0] / 16;  // FIN = 16
  const int E = in_sizes[2];
  const int EP = E + N;

  const float* x   = (const float*)d_in[0];
  const int*   src = (const int*)d_in[2];
  const int*   dst = (const int*)d_in[3];
  const float* W1  = (const float*)d_in[4];  const float* b1  = (const float*)d_in[5];
  const float* W2  = (const float*)d_in[6];  const float* b2  = (const float*)d_in[7];
  const float* Wl1 = (const float*)d_in[8];  const float* bl1 = (const float*)d_in[9];
  const float* Wr1 = (const float*)d_in[10]; const float* br1 = (const float*)d_in[11];
  const float* att1= (const float*)d_in[12]; const float* cb1 = (const float*)d_in[13];
  const float* W4  = (const float*)d_in[14]; const float* b4  = (const float*)d_in[15];
  const float* Wl2 = (const float*)d_in[16]; const float* bl2 = (const float*)d_in[17];
  const float* Wr2 = (const float*)d_in[18]; const float* br2 = (const float*)d_in[19];
  const float* att2= (const float*)d_in[20]; const float* cb2 = (const float*)d_in[21];
  const float* W5  = (const float*)d_in[22]; const float* b5  = (const float*)d_in[23];
  const float* We1 = (const float*)d_in[24]; const float* be1 = (const float*)d_in[25];
  const float* We2 = (const float*)d_in[26]; const float* be2 = (const float*)d_in[27];
  const float* We3 = (const float*)d_in[28]; const float* be3 = (const float*)d_in[29];
  float* out = (float*)d_out;

  // workspace carve-up (256B aligned slices)
  char* ws = (char*)d_ws;
  size_t off = 0;
  auto take = [&](size_t bytes) -> void* {
    void* p = ws + off;
    off = (off + bytes + 255) & ~(size_t)255;
    return p;
  };
  float* hA   = (float*)take((size_t)N * HID * 4);
  float* hB   = (float*)take((size_t)N * HID * 4);
  float* glb  = (float*)take((size_t)N * HID * 4);
  float* grb  = (float*)take((size_t)N * HID * 4);
  float* accb = (float*)take((size_t)N * HID * 4);
  float* mx   = (float*)take((size_t)N * 4 * 4);
  float* sm   = (float*)take((size_t)N * 4 * 4);
  float* ebuf = (float*)take((size_t)EP * 4 * 4);
  // packed weights: KT * 4 * 32 * 16 halves
  _Float16* W1p  = (_Float16*)take(1 * 2048 * 2);
  _Float16* W2p  = (_Float16*)take(2 * 2048 * 2);
  _Float16* Wl1p = (_Float16*)take(2 * 2048 * 2);
  _Float16* Wr1p = (_Float16*)take(2 * 2048 * 2);
  _Float16* W4p  = (_Float16*)take(2 * 2048 * 2);
  _Float16* Wl2p = (_Float16*)take(2 * 2048 * 2);
  _Float16* Wr2p = (_Float16*)take(2 * 2048 * 2);
  _Float16* W5p  = (_Float16*)take(2 * 2048 * 2);
  _Float16* We1p = (_Float16*)take(4 * 2048 * 2);
  _Float16* We2p = (_Float16*)take(2 * 2048 * 2);
  _Float16* We3h = (_Float16*)take(64 * 2);

  auto pack = [&](const float* s, _Float16* d, int K) {
    int total = ((K + 31) >> 5) * 2048;
    k_pack_b<<<(total + 255) / 256, 256, 0, stream>>>(s, d, K);
  };
  pack(W1, W1p, 16);   pack(W2, W2p, 64);
  pack(Wl1, Wl1p, 64); pack(Wr1, Wr1p, 64);
  pack(W4, W4p, 64);
  pack(Wl2, Wl2p, 64); pack(Wr2, Wr2p, 64);
  pack(W5, W5p, 64);
  pack(We1, We1p, 128); pack(We2, We2p, 64);
  k_f32_to_f16<<<1, 64, 0, stream>>>(We3, We3h, 64);

  auto fill = [&](float* p, float v, long long n) {
    int blocks = (int)((n + 255) / 256);
    if (blocks > 16384) blocks = 16384;
    k_fill_f32<<<blocks, 256, 0, stream>>>(p, v, n);
  };

  const int gblocks = (N / 16 + 7) / 8;
  auto gemm_bias = [&](const float* A, const _Float16* Wp, const float* bias,
                       float* o) {  // MODE 0, K=64
    k_gemm_node<64, 0><<<gblocks, 256, 0, stream>>>(A, Wp, bias, nullptr, o, N);
  };
  auto gemm_resid = [&](const float* A, const _Float16* Wp, const float* bias,
                        const float* resid, float* o) {  // MODE 2, K=64
    k_gemm_node<64, 2><<<gblocks, 256, 0, stream>>>(A, Wp, bias, resid, o, N);
  };

  const int ethreads = EP * 4;
  const int eblocks = (ethreads + 255) / 256;
  auto run_gat = [&](const float* hIn, float* hOut, const _Float16* Wlp,
                     const float* bl, const _Float16* Wrp, const float* br,
                     const float* att, const float* cb) {
    gemm_bias(hIn, Wlp, bl, glb);   // gl = h@Wl + bl
    gemm_bias(hIn, Wrp, br, grb);   // gr = h@Wr + br
    fill(mx, -3.0e38f, (long long)N * 4);
    fill(sm, 0.f, (long long)N * 4);
    fill(accb, 0.f, (long long)N * HID);
    k_gat_p1<<<eblocks, 256, 0, stream>>>(glb, grb, src, dst, att, ebuf, mx, E, EP);
    k_gat_p2<<<eblocks, 256, 0, stream>>>(ebuf, dst, mx, sm, E, EP);
    k_gat_p3<<<eblocks, 256, 0, stream>>>(ebuf, glb, src, dst, sm, accb, E, EP);
    k_gat_fin<<<(N * HID + 255) / 256, 256, 0, stream>>>(hIn, accb, cb, hOut, N * HID);
  };

  // h = relu(x @ W1 + b1)
  k_gemm_node<16, 1><<<gblocks, 256, 0, stream>>>(x, W1p, b1, nullptr, hA, N);
  // h = h + relu(h @ W2 + b2)
  gemm_resid(hA, W2p, b2, hA, hB);
  // GAT layer 1: hA = hB + gatv2(hB) + cb1
  run_gat(hB, hA, Wl1p, bl1, Wr1p, br1, att1, cb1);
  // h = h + relu(h @ W4 + b4)
  gemm_resid(hA, W4p, b4, hA, hB);
  // GAT layer 2
  run_gat(hB, hA, Wl2p, bl2, Wr2p, br2, att2, cb2);
  // h = h + relu(h @ W5 + b5)
  gemm_resid(hA, W5p, b5, hA, hB);

  // zero dense output, then fused edge MLP + scatter
  fill(out, 0.f, (long long)N * (long long)N);
  const int waveTiles = (E + 15) / 16;
  const int mblocks = (waveTiles + 7) / 8;
  k_edge_mlp<<<mblocks, 256, 0, stream>>>(hB, src, dst, We1p, be1, We2p, be2,
                                          We3h, be3, out, N, E);
}
